// LLaDA2MoeAttention_86148454023768
// MI455X (gfx1250) — compile-verified, matrix operand order checked by reading, not run on hardware
//
#include <hip/hip_runtime.h>
#include <hip/hip_bf16.h>
#include <math.h>

// ---------------------------------------------------------------------------
// MI455X (gfx1250): bf16 WMMA for all matmuls; async global->LDS DMA
// (ASYNCcnt) triple-buffered with partial s_wait_asynccnt so DMA of tile k+2
// streams while WMMAs run on tile k.
// Pipeline: f32->bf16 convert, QKV GEMM, rmsnorm+rope, flash attention,
// dense GEMM.
// ---------------------------------------------------------------------------

#define B_   2
#define S_   2048
#define HID_ 2048
#define NH_  16
#define NKV_ 4
#define HD_  128
#define RD_  64
#define QKVO_ 3072   // (NH + 2*NKV) * HD

typedef __bf16 bf16_t;
typedef __attribute__((ext_vector_type(16))) __bf16 v16bf;
typedef __attribute__((ext_vector_type(8)))  __bf16 v8bf;
typedef __attribute__((ext_vector_type(8)))  float  v8f;

static __device__ inline v8f zero8() {
  v8f r;
#pragma unroll
  for (int i = 0; i < 8; ++i) r[i] = 0.0f;
  return r;
}

// 16-byte async DMA: global -> LDS, tracked by ASYNCcnt (CDNA5 path).
// LDS operand is the 32-bit LDS address = low 32 bits of the generic pointer
// (ISA: LDS aperture mapping truncates to addr[31:0]).
static __device__ inline void async_cp16(void* lds, const void* gsrc) {
  unsigned loff = (unsigned)(size_t)lds;
  asm volatile("global_load_async_to_lds_b128 %0, %1, off"
               :: "v"(loff), "v"(gsrc)
               : "memory");
}
// Partial wait: ASYNCcnt completes in order per wave, so waiting <=4 after
// issuing one 4-op stage guarantees the *previous* stage has fully landed
// while the newest stage's DMA stays in flight.
static __device__ inline void wait_async_le4() {
  asm volatile("s_wait_asynccnt 0x4" ::: "memory");
}
static __device__ inline void wait_async0() {
  asm volatile("s_wait_asynccnt 0x0" ::: "memory");
}
static __device__ inline void wait_async(bool deep) {
  if (deep) wait_async_le4(); else wait_async0();
}

// Build a 16-element bf16 fragment from two 8-element (16-byte) chunks.
// Matches ISA 16-bit A/B VGPR layout: lane<16 -> K {0..7,16..23},
// lane>=16 -> K {8..15,24..31}.
static __device__ inline v16bf frag16(const bf16_t* p0, const bf16_t* p1) {
  v8bf a = *(const v8bf*)p0;
  v8bf b = *(const v8bf*)p1;
  v16bf r;
#pragma unroll
  for (int i = 0; i < 8; ++i) { r[i] = a[i]; r[i + 8] = b[i]; }
  return r;
}

static __device__ inline float half16_max(float x) {
#pragma unroll
  for (int m = 1; m < 16; m <<= 1) x = fmaxf(x, __shfl_xor(x, m, 32));
  return x;
}
static __device__ inline float half16_sum(float x) {
#pragma unroll
  for (int m = 1; m < 16; m <<= 1) x += __shfl_xor(x, m, 32);
  return x;
}

// ---------------------------------------------------------------------------
__global__ void f32_to_bf16_kernel(const float* __restrict__ x,
                                   bf16_t* __restrict__ y, long n) {
  long i = (long)blockIdx.x * blockDim.x + threadIdx.x;
  long stride = (long)gridDim.x * blockDim.x;
  for (; i < n; i += stride) y[i] = (bf16_t)x[i];
}

// ---------------------------------------------------------------------------
// C[M,N] (fp32) = A[M,K] * B[N,K]^T, bf16 operands, K contiguous.
// 256 threads = 8 waves as 4(M)x2(N); tile 128x128; K in 32-wide steps;
// triple-buffered LDS staged by async DMA, partial-wait pipelined.
__global__ void __launch_bounds__(256)
gemm_bf16_nt_kernel(const bf16_t* __restrict__ A, const bf16_t* __restrict__ Bm,
                    float* __restrict__ C, int M, int N, int K) {
  __shared__ __align__(16) bf16_t as[3][128][40];
  __shared__ __align__(16) bf16_t bs[3][128][40];

  const int tid  = threadIdx.x;
  const int lane = tid & 31;
  const int wv   = tid >> 5;
  const int mw   = wv & 3;           // 4 M-groups of 32 rows
  const int nw   = wv >> 2;          // 2 N-groups of 64 cols
  const int m0   = blockIdx.y * 128;
  const int n0   = blockIdx.x * 128;
  const int mrow = lane & 15;
  const int off  = (lane < 16) ? 0 : 8;   // K-chunk offset per ISA VGPR layout

  // One stage = 4 async instructions per wave (2 for A, 2 for B).
  auto stage = [&](int buf, int k0) {
#pragma unroll
    for (int i = 0; i < 2; ++i) {         // 512 chunks / 256 threads
      int idx = tid + i * 256;
      int r = idx >> 2, cv = idx & 3;
      async_cp16(&as[buf][r][cv * 8], &A[(size_t)(m0 + r) * K + k0 + cv * 8]);
      async_cp16(&bs[buf][r][cv * 8], &Bm[(size_t)(n0 + r) * K + k0 + cv * 8]);
    }
  };

  v8f acc[2][4];
#pragma unroll
  for (int mt = 0; mt < 2; ++mt)
#pragma unroll
    for (int nt = 0; nt < 4; ++nt) acc[mt][nt] = zero8();

  const int nk = K >> 5;
  stage(0, 0);
  if (nk > 1) stage(1, 32);
  wait_async(nk > 1);          // tile 0 landed; tile 1 may still be in flight
  __syncthreads();

  for (int kc = 0; kc < nk; ++kc) {
    const int cur = kc % 3;
    if (kc + 2 < nk) stage((kc + 2) % 3, (kc + 2) * 32);  // overlaps WMMA

    v16bf af[2];
#pragma unroll
    for (int mt = 0; mt < 2; ++mt) {
      const bf16_t* p = &as[cur][mw * 32 + mt * 16 + mrow][0];
      af[mt] = frag16(p + off, p + off + 16);
    }
#pragma unroll
    for (int nt = 0; nt < 4; ++nt) {
      const bf16_t* p = &bs[cur][nw * 64 + nt * 16 + mrow][0];
      v16bf bfr = frag16(p + off, p + off + 16);
#pragma unroll
      for (int mt = 0; mt < 2; ++mt)
        acc[mt][nt] = __builtin_amdgcn_wmma_f32_16x16x32_bf16(
            false, af[mt], false, bfr, (short)0, acc[mt][nt], false, false);
    }

    wait_async(kc + 2 < nk);   // ensure tile kc+1 landed; kc+2 stays in flight
    __syncthreads();
  }

  // C layout: VGPR r -> row (r + off), lane%16 -> column within 16-wide tile.
#pragma unroll
  for (int mt = 0; mt < 2; ++mt)
#pragma unroll
    for (int nt = 0; nt < 4; ++nt)
#pragma unroll
      for (int r = 0; r < 8; ++r)
        C[(size_t)(m0 + mw * 32 + mt * 16 + r + off) * N +
          n0 + nw * 64 + nt * 16 + mrow] = acc[mt][nt][r];
}

// ---------------------------------------------------------------------------
// One wave per (row = b*S+s, head). Heads 0..15: q (rmsnorm+rope),
// 16..19: k (rmsnorm+rope), 20..23: v (passthrough). Output head-major bf16.
__global__ void __launch_bounds__(128)
rmsnorm_rope_kernel(const float* __restrict__ qkv, const float* __restrict__ cosb,
                    const float* __restrict__ sinb, const float* __restrict__ qw,
                    const float* __restrict__ kw, bf16_t* __restrict__ Qo,
                    bf16_t* __restrict__ Ko, bf16_t* __restrict__ Vo) {
  const int tid  = threadIdx.x;
  const int lane = tid & 31;
  const int gw   = blockIdx.x * 4 + (tid >> 5);
  const int head = gw % 24;
  const int row  = gw / 24;            // b*S + s
  const int b    = row >> 11;
  const int s    = row & (S_ - 1);

  const float* x = qkv + (size_t)row * QKVO_ + head * HD_;
  float v0 = x[lane], v1 = x[lane + 32], v2 = x[lane + 64], v3 = x[lane + 96];
  float o0, o1, o2, o3;

  if (head < 20) {
    float ss = v0 * v0 + v1 * v1 + v2 * v2 + v3 * v3;
#pragma unroll
    for (int m = 1; m < 32; m <<= 1) ss += __shfl_xor(ss, m, 32);
    float rs = rsqrtf(ss * (1.0f / 128.0f) + 1e-6f);
    const float* w = (head < 16) ? qw : kw;
    float n0 = v0 * rs * w[lane],      n1 = v1 * rs * w[lane + 32];
    float n2 = v2 * rs * w[lane + 64], n3 = v3 * rs * w[lane + 96];
    const float* cb = cosb + (size_t)row * RD_;
    const float* sb = sinb + (size_t)row * RD_;
    float c0 = cb[lane], c1 = cb[lane + 32];
    float si0 = sb[lane], si1 = sb[lane + 32];
    o0 = n0 * c0 - n1 * si0;   // rotate_half: first half uses -x2*sin
    o1 = n1 * c1 + n0 * si1;   // second half uses +x1*sin
    o2 = n2; o3 = n3;          // dims >= RD pass through
  } else {
    o0 = v0; o1 = v1; o2 = v2; o3 = v3;
  }

  bf16_t* dst;
  if (head < 16)
    dst = Qo + ((size_t)(b * NH_ + head) * S_ + s) * HD_;
  else if (head < 20)
    dst = Ko + ((size_t)(b * NKV_ + (head - 16)) * S_ + s) * HD_;
  else
    dst = Vo + ((size_t)(b * NKV_ + (head - 20)) * S_ + s) * HD_;
  dst[lane] = (bf16_t)o0; dst[lane + 32] = (bf16_t)o1;
  dst[lane + 64] = (bf16_t)o2; dst[lane + 96] = (bf16_t)o3;
}

// ---------------------------------------------------------------------------
// Flash attention: workgroup = (b, h, 64 q-rows); 4 waves, 16 q-rows each.
// kv in blocks of 32: QK^T (8 wmma), online softmax, PV (8 wmma).
// K tile: triple-buffered async DMA; V tile (transposed): register pipeline.
__global__ void __launch_bounds__(128)
attention_kernel(const bf16_t* __restrict__ Qg, const bf16_t* __restrict__ Kg,
                 const bf16_t* __restrict__ Vg, bf16_t* __restrict__ Og) {
  __shared__ __align__(16) bf16_t kt[3][32][136];   // K tile  [kv][d]
  __shared__ __align__(16) bf16_t vt[2][128][40];   // V tile  [d][kv] transposed
  __shared__ __align__(16) bf16_t ps[4][16][40];    // per-wave P scratch

  const int tid  = threadIdx.x;
  const int lane = tid & 31;
  const int wv   = tid >> 5;
  const int bid  = blockIdx.x;
  const int qt   = bid & 31;             // S/64 = 32 q-tiles
  const int h    = (bid >> 5) % NH_;
  const int b    = bid / (32 * NH_);
  const int kvh  = h / (NH_ / NKV_);
  const int q0   = qt * 64;
  const int mrow = lane & 15;
  const int off  = (lane < 16) ? 0 : 8;
  const float scale = 0.08838834764831845f;   // 1/sqrt(128)

  const bf16_t* Kbase = Kg + (size_t)(b * NKV_ + kvh) * S_ * HD_;
  const bf16_t* Vbase = Vg + (size_t)(b * NKV_ + kvh) * S_ * HD_;

  // This wave's 16x128 Q block as 4 A-fragments (held for whole kernel).
  const bf16_t* Qrow =
      Qg + ((size_t)(b * NH_ + h) * S_ + q0 + wv * 16 + mrow) * HD_;
  v16bf qa[4];
#pragma unroll
  for (int c = 0; c < 4; ++c)
    qa[c] = frag16(Qrow + c * 32 + off, Qrow + c * 32 + off + 16);

  v8f o_acc[8];
#pragma unroll
  for (int t = 0; t < 8; ++t) o_acc[t] = zero8();
  float m_i[8], l_i[8];
#pragma unroll
  for (int r = 0; r < 8; ++r) { m_i[r] = -1e30f; l_i[r] = 0.0f; }

  v8bf vreg[4];
  auto stageK = [&](int buf, int blk) {      // 4 async instructions per wave
#pragma unroll
    for (int i = 0; i < 4; ++i) {
      int idx = tid + i * 128;
      int kv = idx >> 4, cv = idx & 15;
      async_cp16(&kt[buf][kv][cv * 8],
                 &Kbase[(size_t)(blk * 32 + kv) * HD_ + cv * 8]);
    }
  };
  auto loadV = [&](int blk) {                // global loads into registers
#pragma unroll
    for (int i = 0; i < 4; ++i) {
      int idx = tid + i * 128;
      int kv = idx >> 4, cv = idx & 15;
      vreg[i] = *(const v8bf*)&Vbase[(size_t)(blk * 32 + kv) * HD_ + cv * 8];
    }
  };
  auto storeV = [&](int buf) {               // transpose into LDS [d][kv]
#pragma unroll
    for (int i = 0; i < 4; ++i) {
      int idx = tid + i * 128;
      int kv = idx >> 4, cv = idx & 15;
#pragma unroll
      for (int j = 0; j < 8; ++j) vt[buf][cv * 8 + j][kv] = vreg[i][j];
    }
  };

  const int nb = S_ / 32;
  stageK(0, 0);
  stageK(1, 1);
  loadV(0);
  storeV(0);
  wait_async_le4();            // kt[0] landed; kt[1] may still be in flight
  __syncthreads();

  for (int blk = 0; blk < nb; ++blk) {
    const int cur = blk % 3;
    if (blk + 2 < nb) stageK((blk + 2) % 3, blk + 2);  // overlaps compute
    if (blk + 1 < nb) loadV(blk + 1);

    // Scores: two 16x16 tiles (kv 0..15 and 16..31), K accumulated over HD.
    v8f s0 = zero8(), s1 = zero8();
#pragma unroll
    for (int c = 0; c < 4; ++c) {
      v16bf b0 = frag16(&kt[cur][mrow][c * 32 + off],
                        &kt[cur][mrow][c * 32 + off + 16]);
      v16bf b1 = frag16(&kt[cur][16 + mrow][c * 32 + off],
                        &kt[cur][16 + mrow][c * 32 + off + 16]);
      s0 = __builtin_amdgcn_wmma_f32_16x16x32_bf16(false, qa[c], false, b0,
                                                   (short)0, s0, false, false);
      s1 = __builtin_amdgcn_wmma_f32_16x16x32_bf16(false, qa[c], false, b1,
                                                   (short)0, s1, false, false);
    }

    // Online softmax (row stats replicated across each 16-lane half).
    float alpha[8];
#pragma unroll
    for (int r = 0; r < 8; ++r) {
      s0[r] *= scale; s1[r] *= scale;
      float mx = half16_max(fmaxf(s0[r], s1[r]));
      float mn = fmaxf(m_i[r], mx);
      alpha[r] = __expf(m_i[r] - mn);
      m_i[r] = mn;
    }
#pragma unroll
    for (int r = 0; r < 8; ++r) {
      float p0 = __expf(s0[r] - m_i[r]);
      float p1 = __expf(s1[r] - m_i[r]);
      s0[r] = p0; s1[r] = p1;
      l_i[r] = l_i[r] * alpha[r] + half16_sum(p0 + p1);
    }
#pragma unroll
    for (int t = 0; t < 8; ++t)
#pragma unroll
      for (int r = 0; r < 8; ++r) o_acc[t][r] *= alpha[r];

    // C-layout -> A-layout transpose of P via per-wave LDS scratch.
#pragma unroll
    for (int r = 0; r < 8; ++r) {
      ps[wv][r + off][mrow]      = (bf16_t)s0[r];
      ps[wv][r + off][16 + mrow] = (bf16_t)s1[r];
    }
    v16bf pa = frag16(&ps[wv][mrow][off], &ps[wv][mrow][off + 16]);

    // PV: 8 output d-tiles of 16 columns each.
#pragma unroll
    for (int t = 0; t < 8; ++t) {
      v16bf vb = frag16(&vt[blk & 1][t * 16 + mrow][off],
                        &vt[blk & 1][t * 16 + mrow][off + 16]);
      o_acc[t] = __builtin_amdgcn_wmma_f32_16x16x32_bf16(
          false, pa, false, vb, (short)0, o_acc[t], false, false);
    }

    if (blk + 1 < nb) storeV((blk + 1) & 1);  // waits vreg loads; after compute
    wait_async(blk + 2 < nb);  // kt[blk+1] landed; kt[blk+2] stays in flight
    __syncthreads();
  }

  // Epilogue: normalize by l and store [b, s, h*HD + d] bf16 for dense GEMM.
#pragma unroll
  for (int t = 0; t < 8; ++t)
#pragma unroll
    for (int r = 0; r < 8; ++r) {
      int srow = q0 + wv * 16 + r + off;
      int d = t * 16 + mrow;
      Og[((size_t)(b * S_ + srow)) * (NH_ * HD_) + h * HD_ + d] =
          (bf16_t)(o_acc[t][r] / l_i[r]);
    }
}

// ---------------------------------------------------------------------------
extern "C" void kernel_launch(void* const* d_in, const int* in_sizes, int n_in,
                              void* d_out, int out_size, void* d_ws, size_t ws_size,
                              hipStream_t stream) {
  const float* hs   = (const float*)d_in[0];   // [B,S,HID]
  const float* cosb = (const float*)d_in[1];   // [B,S,RD]
  const float* sinb = (const float*)d_in[2];   // [B,S,RD]
  const float* wqkv = (const float*)d_in[3];   // [3072, 2048]
  const float* qw   = (const float*)d_in[4];   // [128]
  const float* kw   = (const float*)d_in[5];   // [128]
  const float* wdns = (const float*)d_in[6];   // [2048, 2048]
  float* out = (float*)d_out;                  // [B,S,2048]

  char* ws = (char*)d_ws;
  size_t off = 0;
  auto alloc = [&](size_t bytes) { char* p = ws + off; off += (bytes + 255) & ~(size_t)255; return p; };

  const long nHS = (long)B_ * S_ * HID_;          // 8388608
  const long nWQ = (long)QKVO_ * HID_;            // 6291456
  const long nWD = (long)HID_ * (NH_ * HD_);      // 4194304

  bf16_t* hsb  = (bf16_t*)alloc(nHS * 2);
  bf16_t* wqb  = (bf16_t*)alloc(nWQ * 2);
  bf16_t* wdb  = (bf16_t*)alloc(nWD * 2);
  float*  qkvf = (float*)alloc((size_t)B_ * S_ * QKVO_ * 4);
  bf16_t* qb   = (bf16_t*)alloc((size_t)B_ * NH_  * S_ * HD_ * 2);
  bf16_t* kb   = (bf16_t*)alloc((size_t)B_ * NKV_ * S_ * HD_ * 2);
  bf16_t* vb   = (bf16_t*)alloc((size_t)B_ * NKV_ * S_ * HD_ * 2);
  bf16_t* attb = (bf16_t*)alloc((size_t)B_ * S_ * NH_ * HD_ * 2);
  (void)ws_size; (void)n_in; (void)in_sizes; (void)out_size;

  f32_to_bf16_kernel<<<4096, 256, 0, stream>>>(hs, hsb, nHS);
  f32_to_bf16_kernel<<<4096, 256, 0, stream>>>(wqkv, wqb, nWQ);
  f32_to_bf16_kernel<<<4096, 256, 0, stream>>>(wdns, wdb, nWD);

  // QKV projection: [4096,2048] x [3072,2048]^T -> fp32
  gemm_bf16_nt_kernel<<<dim3(QKVO_ / 128, (B_ * S_) / 128), 256, 0, stream>>>(
      hsb, wqb, qkvf, B_ * S_, QKVO_, HID_);

  // RMSNorm + RoPE + head-major layout
  rmsnorm_rope_kernel<<<(B_ * S_ * 24) / 4, 128, 0, stream>>>(
      qkvf, cosb, sinb, qw, kw, qb, kb, vb);

  // Flash attention: (b, h, q-tile) grid
  attention_kernel<<<B_ * NH_ * (S_ / 64), 128, 0, stream>>>(qb, kb, vb, attb);

  // Output projection: [4096,2048] x [2048,2048]^T -> fp32 out
  gemm_bf16_nt_kernel<<<dim3((NH_ * HD_) / 128, (B_ * S_) / 128), 256, 0, stream>>>(
      attb, wdb, out, B_ * S_, NH_ * HD_, HID_);
}
